// HeteroGNN_87368224735716
// MI455X (gfx1250) — matrix operand order repaired and checked by reading, not compiled, not run on hardware
//
#include <hip/hip_runtime.h>
#include <math.h>

typedef __attribute__((ext_vector_type(16))) __bf16 v16bf;
typedef __attribute__((ext_vector_type(8)))  __bf16 v8bf;
typedef __attribute__((ext_vector_type(8)))  float  v8f;

#define HDIM 128
#define LDR  136   // LDS row pitch (elements): 272B, 16B-aligned, bank-conflict-free
#define NEG_SLOPE 0.2f

union bfrag { v16bf v; v8bf h[2]; };

// ---------------------------------------------------------------------------
// WMMA GEMM: Y[nrows,128] = X[nrows,128] @ W[128,128]  (bf16 mul, fp32 acc)
// Block: 256 threads = 8 waves. Block owns 16 rows; wave w owns cols [16w,16w+16).
// W is staged transposed in LDS so both A and B fragments are contiguous
// 16-byte runs per lane -> ds_load_b128 for every fragment piece.
// ---------------------------------------------------------------------------
__global__ __launch_bounds__(256) void gemm_bf16_wmma_kernel(
    const float* __restrict__ X, const float* __restrict__ W,
    float* __restrict__ Y, int nrows)
{
  __shared__ __bf16 sX[16 * LDR];          // ~4.3 KB, row-major rows of X tile
  __shared__ __bf16 sWt[HDIM * LDR];       // ~34 KB, W transposed: sWt[n][k]
  const int tid  = threadIdx.x;
  const int wave = tid >> 5;
  const int lane = tid & 31;
  const int row0 = blockIdx.x << 4;

  // Stage W transposed (coalesced global reads; one-time strided LDS writes).
  for (int i = tid; i < HDIM * HDIM; i += 256) {
    const int k = i >> 7, n = i & (HDIM - 1);
    sWt[n * LDR + k] = (__bf16)W[i];
  }
  // Stage X tile (row-major).
  for (int i = tid; i < 16 * HDIM; i += 256) {
    const int r = i >> 7, c = i & (HDIM - 1);
    const int gr = row0 + r;
    sX[r * LDR + c] = (__bf16)((gr < nrows) ? X[(size_t)gr * HDIM + c] : 0.0f);
  }
  __syncthreads();

  const int half = lane >> 4;       // 0: lanes 0-15, 1: lanes 16-31
  const int m    = lane & 15;       // A row / D col within tile
  const int ncol = (wave << 4) + m; // global output column
  v8f acc = {};

#pragma unroll
  for (int k0 = 0; k0 < HDIM; k0 += 32) {
    // A 16x32 bf16 (ISA 7.12.2): lane(m,half) elems 0-7 = K k0+8*half+0..7,
    // elems 8-15 = K k0+16+8*half+0..7  -> two contiguous 16B runs.
    bfrag a, b;
    const __bf16* ax = &sX[m * LDR + k0 + half * 8];
    a.h[0] = *reinterpret_cast<const v8bf*>(ax);
    a.h[1] = *reinterpret_cast<const v8bf*>(ax + 16);
    // B 32x16: lane holds column (lane&15); same K order as A -> two 16B runs
    // from the transposed slab.
    const __bf16* bx = &sWt[ncol * LDR + k0 + half * 8];
    b.h[0] = *reinterpret_cast<const v8bf*>(bx);
    b.h[1] = *reinterpret_cast<const v8bf*>(bx + 16);
    acc = __builtin_amdgcn_wmma_f32_16x16x32_bf16(false, a.v, false, b.v,
                                                  (short)0, acc, false, false);
  }

  // D layout: acc[r] -> row (r + 8*half), col (lane&15).
  // Uniform fast path: full tiles (the common case) store unguarded, so the
  // compiler emits 8 plain global_store_b32 without per-lane exec juggling.
  if (row0 + 16 <= nrows) {
    float* yp = Y + (size_t)(row0 + 8 * half) * HDIM + ncol;
#pragma unroll
    for (int r = 0; r < 8; ++r) yp[(size_t)r * HDIM] = acc[r];
  } else {
#pragma unroll
    for (int r = 0; r < 8; ++r) {
      const int grow = row0 + r + 8 * half;
      if (grow < nrows) Y[(size_t)grow * HDIM + ncol] = acc[r];
    }
  }
}

// ---------------------------------------------------------------------------
// a[i] = dot(Hm[i,:], att)  — one wave per row, shuffle reduction (wave32)
// ---------------------------------------------------------------------------
__global__ __launch_bounds__(256) void rowdot_kernel(
    const float* __restrict__ Hm, const float* __restrict__ att,
    float* __restrict__ a, int n)
{
  const int row  = blockIdx.x * 8 + (threadIdx.x >> 5);
  const int lane = threadIdx.x & 31;
  if (row >= n) return;
  const float* r = Hm + (size_t)row * HDIM;
  float s = r[lane]      * att[lane]
          + r[lane + 32] * att[lane + 32]
          + r[lane + 64] * att[lane + 64]
          + r[lane + 96] * att[lane + 96];
  for (int off = 16; off > 0; off >>= 1) s += __shfl_down(s, off, 32);
  if (lane == 0) a[row] = s;
}

__device__ __forceinline__ float leaky(float v) { return v > 0.f ? v : NEG_SLOPE * v; }

__device__ __forceinline__ void atomicMaxF(float* addr, float val) {
  unsigned int* ua = reinterpret_cast<unsigned int*>(addr);
  unsigned int old = *ua;
  while (__uint_as_float(old) < val) {
    const unsigned int assumed = old;
    old = atomicCAS(ua, assumed, __float_as_uint(val));
    if (old == assumed) break;
  }
}

__global__ void fill_kernel(float* __restrict__ p, float v, int n) {
  const int i = blockIdx.x * blockDim.x + threadIdx.x;
  if (i < n) p[i] = v;
}

__global__ void init_bias_kernel(float* __restrict__ out, const float* __restrict__ bias, int n) {
  const int i = blockIdx.x * blockDim.x + threadIdx.x;
  if (i < n) out[i] = bias[i & (HDIM - 1)];
}

__global__ void edge_max_kernel(const float* __restrict__ as, const float* __restrict__ ad,
                                const int* __restrict__ src, const int* __restrict__ dst,
                                float* __restrict__ m, int E_) {
  const int e = blockIdx.x * blockDim.x + threadIdx.x;
  if (e >= E_) return;
  const int d = dst[e];
  atomicMaxF(&m[d], leaky(as[src[e]] + ad[d]));
}

__global__ void edge_exp_kernel(const float* __restrict__ as, const float* __restrict__ ad,
                                const int* __restrict__ src, const int* __restrict__ dst,
                                const float* __restrict__ m, float* __restrict__ den,
                                float* __restrict__ ex, int E_) {
  const int e = blockIdx.x * blockDim.x + threadIdx.x;
  if (e >= E_) return;
  const int d = dst[e];
  const float x = __expf(leaky(as[src[e]] + ad[d]) - m[d]);
  ex[e] = x;
  atomicAdd(&den[d], x);
}

// One wave per edge: lane l moves cols [4l,4l+4) of hs[src] -> out[dst] (float4 loads)
__global__ __launch_bounds__(256) void edge_scatter_kernel(
    const float* __restrict__ hs, const int* __restrict__ src, const int* __restrict__ dst,
    const float* __restrict__ ex, const float* __restrict__ den,
    float* __restrict__ out, int E_) {
  const int e = blockIdx.x * 8 + (threadIdx.x >> 5);
  if (e >= E_) return;
  const int lane = threadIdx.x & 31;
  const int s = src[e], d = dst[e];
  const float w = ex[e] / den[d];
  const float4 v = reinterpret_cast<const float4*>(hs + (size_t)s * HDIM)[lane];
  float* orow = out + (size_t)d * HDIM + lane * 4;
  atomicAdd(orow + 0, w * v.x);
  atomicAdd(orow + 1, w * v.y);
  atomicAdd(orow + 2, w * v.z);
  atomicAdd(orow + 3, w * v.w);
}

__global__ void relu_kernel(const float* __restrict__ in, float* __restrict__ out, int n) {
  const int i = blockIdx.x * blockDim.x + threadIdx.x;
  if (i < n) out[i] = fmaxf(in[i], 0.0f);
}

__global__ void pool_max_kernel(const float* __restrict__ x, const int* __restrict__ batch,
                                float* __restrict__ pooled, int n) {
  const int i = blockIdx.x * blockDim.x + threadIdx.x;
  if (i >= n) return;
  const int node = i >> 7, c = i & (HDIM - 1);
  atomicMaxF(&pooled[(size_t)batch[node] * HDIM + c], x[i]);
}

__global__ void final_linear_kernel(const float* __restrict__ pooledP,
                                    const float* __restrict__ pooledA,
                                    const float* __restrict__ linW,
                                    const float* __restrict__ linb,
                                    float* __restrict__ out, int G_, int OUT_) {
  const int t = blockIdx.x * blockDim.x + threadIdx.x;
  const int total = 2 * G_ * OUT_;
  if (t >= total) return;
  const int which = t / (G_ * OUT_);
  const int rem = t % (G_ * OUT_);
  const int g = rem / OUT_, o = rem % OUT_;
  const float* pooled = which ? pooledA : pooledP;
  float s = linb[o];
  for (int k = 0; k < HDIM; ++k) s += pooled[(size_t)g * HDIM + k] * linW[k * OUT_ + o];
  out[t] = s;
}

// ---------------------------------------------------------------------------
extern "C" void kernel_launch(void* const* d_in, const int* in_sizes, int n_in,
                              void* d_out, int out_size, void* d_ws, size_t ws_size,
                              hipStream_t stream) {
  (void)n_in; (void)ws_size;
  const float* xp   = (const float*)d_in[0];
  const float* xa   = (const float*)d_in[1];
  const int*   e_pa = (const int*)d_in[2];
  const int*   e_ap = (const int*)d_in[3];
  const int*   bpap = (const int*)d_in[4];
  const int*   baut = (const int*)d_in[5];
  const float* Wsrc = (const float*)d_in[6];
  const float* Wdst = (const float*)d_in[7];
  const float* attS = (const float*)d_in[8];
  const float* attD = (const float*)d_in[9];
  const float* bias = (const float*)d_in[10];
  const float* linW = (const float*)d_in[11];
  const float* linb = (const float*)d_in[12];

  const int N   = in_sizes[0] / HDIM;
  const int E   = in_sizes[2] / 2;
  const int L   = in_sizes[6] / (2 * HDIM * HDIM);
  const int OUT = in_sizes[12];
  const int G   = out_size / (2 * OUT);

  float* ws = (float*)d_ws;
  const size_t NH = (size_t)N * HDIM;
  float* xp_cur  = ws;
  float* xa_cur  = ws + NH;
  float* hsb     = ws + 2 * NH;
  float* hdb     = ws + 3 * NH;
  float* out_a   = ws + 4 * NH;   // edge type 0 (paper->author) aggregates to authors
  float* out_p   = ws + 5 * NH;   // edge type 1 (author->paper) aggregates to papers
  float* a_s     = ws + 6 * NH;
  float* a_d     = a_s + N;
  float* mm      = a_d + N;
  float* den     = mm + N;
  float* ex      = den + N;
  float* pooledP = ex + E;
  float* pooledA = pooledP + (size_t)G * HDIM;

  const int gemmGrid = (N + 15) / 16;
  const int nGrid    = (N + 255) / 256;
  const int nhGrid   = ((int)NH + 255) / 256;
  const int eGrid    = (E + 255) / 256;

  for (int l = 0; l < L; ++l) {
    const float* cxp = (l == 0) ? xp : xp_cur;
    const float* cxa = (l == 0) ? xa : xa_cur;
    for (int t = 0; t < 2; ++t) {
      const float* xs   = (t == 0) ? cxp : cxa;
      const float* xd   = (t == 0) ? cxa : cxp;
      const int*   edge = (t == 0) ? e_pa : e_ap;
      const int*   src  = edge;
      const int*   dst  = edge + E;
      float*       outb = (t == 0) ? out_a : out_p;
      const size_t po   = (size_t)l * 2 + t;

      gemm_bf16_wmma_kernel<<<gemmGrid, 256, 0, stream>>>(xs, Wsrc + po * HDIM * HDIM, hsb, N);
      gemm_bf16_wmma_kernel<<<gemmGrid, 256, 0, stream>>>(xd, Wdst + po * HDIM * HDIM, hdb, N);
      rowdot_kernel<<<(N + 7) / 8, 256, 0, stream>>>(hsb, attS + po * HDIM, a_s, N);
      rowdot_kernel<<<(N + 7) / 8, 256, 0, stream>>>(hdb, attD + po * HDIM, a_d, N);
      fill_kernel<<<nGrid, 256, 0, stream>>>(mm, -INFINITY, N);
      fill_kernel<<<nGrid, 256, 0, stream>>>(den, 1e-16f, N);
      init_bias_kernel<<<nhGrid, 256, 0, stream>>>(outb, bias + po * HDIM, (int)NH);
      edge_max_kernel<<<eGrid, 256, 0, stream>>>(a_s, a_d, src, dst, mm, E);
      edge_exp_kernel<<<eGrid, 256, 0, stream>>>(a_s, a_d, src, dst, mm, den, ex, E);
      edge_scatter_kernel<<<(E + 7) / 8, 256, 0, stream>>>(hsb, src, dst, ex, den, outb, E);
    }
    relu_kernel<<<nhGrid, 256, 0, stream>>>(out_p, xp_cur, (int)NH);
    relu_kernel<<<nhGrid, 256, 0, stream>>>(out_a, xa_cur, (int)NH);
  }

  const int pGrid = (G * HDIM + 255) / 256;
  fill_kernel<<<pGrid, 256, 0, stream>>>(pooledP, -INFINITY, G * HDIM);
  fill_kernel<<<pGrid, 256, 0, stream>>>(pooledA, -INFINITY, G * HDIM);
  pool_max_kernel<<<nhGrid, 256, 0, stream>>>(xp_cur, bpap, pooledP, (int)NH);
  pool_max_kernel<<<nhGrid, 256, 0, stream>>>(xa_cur, baut, pooledA, (int)NH);
  final_linear_kernel<<<(2 * G * OUT + 255) / 256, 256, 0, stream>>>(
      pooledP, pooledA, linW, linb, (float*)d_out, G, OUT);
}
